// Actor_attf_single_66975720013849
// MI455X (gfx1250) — matrix operand order repaired and checked by reading, not compiled, not run on hardware
//
#include <hip/hip_runtime.h>
#include <hip/hip_bf16.h>
#include <stdint.h>

typedef _Float16 half_t;
typedef __attribute__((ext_vector_type(16))) _Float16 v16h;
typedef __attribute__((ext_vector_type(2)))  _Float16 v2h;
typedef __attribute__((ext_vector_type(8)))  float    v8f;
typedef __attribute__((ext_vector_type(8)))  uint32_t v8u;

// may_alias types: LDS tiles are written as f16/f32 and read back as u32
// vectors; TBAA must NOT be allowed to reorder those accesses (correctness
// relies on the in-order LDS pipe + preserved program order).
typedef __attribute__((ext_vector_type(4))) uint32_t v4u_;
typedef v4u_  __attribute__((may_alias)) v4ua;
typedef __attribute__((ext_vector_type(4))) float v4f_;
typedef v4f_  __attribute__((may_alias)) v4fa;
typedef __attribute__((ext_vector_type(2))) float v2f_;
typedef v2f_  __attribute__((may_alias)) v2fa;
typedef uint32_t __attribute__((may_alias)) u32a;
typedef _Float16 __attribute__((may_alias)) h16a;

#define WAVES_PER_BLOCK 4
// 6144B input tile + 1024B h0 + 1024B h1 + 2048B merged
#define WAVE_SMEM 10240

// compiler-only fence (zero instructions): pins memory ops at stage edges.
__device__ __forceinline__ void comp_fence() { asm volatile("" ::: "memory"); }

__device__ __forceinline__ v8f wmma_f16(v16h a, v16h b, v8f c) {
  return __builtin_amdgcn_wmma_f32_16x16x32_f16(false, a, false, b, (short)0, c,
                                                false, false);
}

// B fragment: lane-major pack -> two coalesced 16B global loads per lane.
__device__ __forceinline__ v16h load_bfrag(const uint32_t* pk, int frag, int lane) {
  const v4ua* base = (const v4ua*)(pk + frag * 256 + lane * 8);
  v4ua a = base[0], b = base[1];
  v8u u;
#pragma unroll
  for (int j = 0; j < 4; ++j) { u[j] = a[j]; u[j + 4] = b[j]; }
  return __builtin_bit_cast(v16h, u);
}

// A fragment (16x32 f16) from LDS tile, row stride S halves: this file's A k order
// (k = hi*16 + s) is contiguous per lane -> two ds_load_b128.
__device__ __forceinline__ v16h load_afrag(const half_t* X, int S, int lane) {
  int m = lane & 15, hi = lane >> 4;
  const u32a* row = (const u32a*)(X + m * S);
  v4ua a = *(const v4ua*)(row + hi * 8);      // K = hi*16 + 0..7
  v4ua b = *(const v4ua*)(row + 4 + hi * 8);  // K = hi*16 + 8..15
  v8u u;
#pragma unroll
  for (int j = 0; j < 4; ++j) { u[j] = a[j]; u[j + 4] = b[j]; }
  return __builtin_bit_cast(v16h, u);
}

// Layer-1 A fragment (K<=4): branchless, only hi==0 lanes carry data.
__device__ __forceinline__ v16h make_a1(float x0, float x1, float x2, float x3, int hi) {
  v2h t0 = {(_Float16)x0, (_Float16)x1};
  v2h t1 = {(_Float16)x2, (_Float16)x3};
  uint32_t p0 = __builtin_bit_cast(uint32_t, t0);
  uint32_t p1 = __builtin_bit_cast(uint32_t, t1);
  v8u u = {};
  u[0] = hi ? 0u : p0;
  u[1] = hi ? 0u : p1;
  return __builtin_bit_cast(v16h, u);
}

// C frag (vgpr r = row r+hi*8, lane&15 = col) -> bias + (leaky)relu -> f16 LDS
__device__ __forceinline__ void store_act(half_t* Y, int S, int noff, v8f c,
                                          float bias, bool leaky, int lane) {
  int nn = (lane & 15) + noff, hi = lane >> 4;
  h16a* y = (h16a*)Y;
#pragma unroll
  for (int r = 0; r < 8; ++r) {
    float v = c[r] + bias;
    v = v > 0.f ? v : (leaky ? 0.01f * v : 0.f);
    y[(r + hi * 8) * S + nn] = (half_t)v;
  }
}

__device__ __forceinline__ float redsum16(float v) {
  v += __shfl_xor(v, 1, 32);
  v += __shfl_xor(v, 2, 32);
  v += __shfl_xor(v, 4, 32);
  v += __shfl_xor(v, 8, 32);
  return v;
}

// Attention pool with 2-way agent pipelining over double-buffered h tiles.
// Plain (unshifted) softmax: scores are O(few), exp cannot overflow, and
// softmax is shift-invariant, so this matches the reference math.
template <int NAG, bool IS_OA>
__device__ __forceinline__ void attend_pool(const float* in_tile, half_t* h0,
                                            half_t* h1, half_t* merged, int lane,
                                            const uint32_t* pk, int fW1,
                                            float b1a, float b1b, float b2n,
                                            float lng, float lnb, int coloff,
                                            const v8f& sc) {
  int m = lane & 15, hi = lane >> 4, n = m;
  v16h w1lo = load_bfrag(pk, fW1 + 0, lane);
  v16h w1hi = load_bfrag(pk, fW1 + 1, lane);
  v16h w2   = load_bfrag(pk, fW1 + 2, lane);
  v8f z = {};
  v8f scq;
  float Ss[8], Pp[8];
#pragma unroll
  for (int r = 0; r < 8; ++r) { Ss[r] = 0.f; Pp[r] = 0.f; scq[r] = sc[r] * 0.25f; }

  auto enc_agent = [&](int k, half_t* hb) -> v8f {
    float x0, x1, x2, x3;
    if constexpr (IS_OA) {
      v2fa p = *(const v2fa*)(in_tile + m * 96 + 4 + 2 * k);
      v2fa q = *(const v2fa*)(in_tile + m * 96 + 34 + 2 * k);
      x0 = p[0]; x1 = p[1]; x2 = q[0]; x3 = q[1];
    } else {
      v2fa p = *(const v2fa*)(in_tile + m * 96 + 64 + 2 * k);
      x0 = p[0]; x1 = p[1]; x2 = 0.f; x3 = 0.f;
    }
    v16h a1 = make_a1(x0, x1, x2, x3, hi);
    v8f c0 = wmma_f16(a1, w1lo, z);
    v8f c1 = wmma_f16(a1, w1hi, z);
    store_act(hb, 32, 0,  c0, b1a, false, lane);
    store_act(hb, 32, 16, c1, b1b, false, lane);
    v16h a2 = load_afrag(hb, 32, lane);  // in-order LDS pipe: sees stores
    v8f e = wmma_f16(a2, w2, z);
#pragma unroll
    for (int r = 0; r < 8; ++r) e[r] = fmaxf(e[r] + b2n, 0.f);  // relu(enc)
    return e;
  };

  auto update = [&](v8f ev) {
#pragma unroll
    for (int r = 0; r < 8; ++r) {
      float s = redsum16(scq[r] * ev[r]);   // scaled score, replicated in-half
      float w = __expf(s);
      Ss[r] += w;
      Pp[r] += w * ev[r];
    }
  };

  int k = 0;
  for (; k + 2 <= NAG; k += 2) {   // two independent encoder chains in flight
    v8f e0 = enc_agent(k, h0);
    v8f e1 = enc_agent(k + 1, h1);
    update(e0);
    update(e1);
  }
  if (k < NAG) update(enc_agent(k, h0));

#pragma unroll
  for (int r = 0; r < 8; ++r) {
    float pool = Pp[r] / Ss[r];
    float mean = redsum16(pool) * 0.0625f;
    float dv   = pool - mean;
    float var  = redsum16(dv * dv) * 0.0625f;
    float y    = fmaxf(dv * rsqrtf(var + 1e-5f) * lng + lnb, 0.f);
    ((h16a*)merged)[(r + hi * 8) * 64 + n + coloff] = (half_t)y;
  }
}

// ---------------------------------------------------------------------------
// Weight pack: 16 B-fragments (32x16 f16). Lane-major: fragment f, lane l,
// dword j at pk[f*256 + l*8 + j]; dword j holds the f16 pair for
// (k = ko + hi*16 + 2j {,+1}, n = (l&15) + no); OOB -> 0.
// ---------------------------------------------------------------------------
__global__ void pack_weights(const float* __restrict__ enW1, const float* __restrict__ enW2,
                             const float* __restrict__ oaW1, const float* __restrict__ oaW2,
                             const float* __restrict__ gW1,  const float* __restrict__ gW2,
                             const float* __restrict__ aW1,  const float* __restrict__ aW2,
                             const float* __restrict__ aW3,  uint32_t* __restrict__ pk) {
  int f = blockIdx.x, lane = threadIdx.x;
  const float* W; int Ks, Ns, ko = 0, no = 0;
  switch (f) {
    case 0:  W = enW1; Ks = 4;  Ns = 32;          break;
    case 1:  W = enW1; Ks = 4;  Ns = 32; no = 16; break;
    case 2:  W = enW2; Ks = 32; Ns = 16;          break;
    case 3:  W = oaW1; Ks = 4;  Ns = 32;          break;
    case 4:  W = oaW1; Ks = 4;  Ns = 32; no = 16; break;
    case 5:  W = oaW2; Ks = 32; Ns = 16;          break;
    case 6:  W = gW1;  Ks = 2;  Ns = 32;          break;
    case 7:  W = gW1;  Ks = 2;  Ns = 32; no = 16; break;
    case 8:  W = gW2;  Ks = 32; Ns = 16;          break;
    case 9:  W = aW1;  Ks = 48; Ns = 32;          break;
    case 10: W = aW1;  Ks = 48; Ns = 32; no = 16; break;
    case 11: W = aW1;  Ks = 48; Ns = 32; ko = 32; break;
    case 12: W = aW1;  Ks = 48; Ns = 32; ko = 32; no = 16; break;
    case 13: W = aW2;  Ks = 32; Ns = 32;          break;
    case 14: W = aW2;  Ks = 32; Ns = 32; no = 16; break;
    default: W = aW3;  Ks = 32; Ns = 2;           break;
  }
  int nn = (lane & 15) + no, hi = lane >> 4;
  uint32_t* out = pk + f * 256 + lane * 8;
#pragma unroll
  for (int j = 0; j < 8; ++j) {
    int k0 = ko + hi * 16 + j * 2;
    _Float16 h0 = (k0 < Ks && nn < Ns) ? (_Float16)W[k0 * Ns + nn] : (_Float16)0.f;
    _Float16 h1 = (k0 + 1 < Ks && nn < Ns) ? (_Float16)W[(k0 + 1) * Ns + nn] : (_Float16)0.f;
    v2h t = {h0, h1};
    out[j] = __builtin_bit_cast(uint32_t, t);
  }
}

// ---------------------------------------------------------------------------
// Main kernel: 128 threads = 4 waves; each wave owns a 16-row batch tile.
// ---------------------------------------------------------------------------
__global__ __launch_bounds__(128) void actor_kernel(
    const float* __restrict__ s_input, const uint32_t* __restrict__ pk,
    const float* __restrict__ en_b1, const float* __restrict__ en_b2,
    const float* __restrict__ oa_b1, const float* __restrict__ oa_b2,
    const float* __restrict__ g_b1,  const float* __restrict__ g_b2,
    const float* __restrict__ oa_g,  const float* __restrict__ oa_bb,
    const float* __restrict__ gl_g,  const float* __restrict__ gl_bb,
    const float* __restrict__ a_b1,  const float* __restrict__ a_b2,
    const float* __restrict__ a_b3,  float* __restrict__ out) {
  __shared__ __align__(16) unsigned char smem[WAVES_PER_BLOCK * WAVE_SMEM];
  const int wave = threadIdx.x >> 5, lane = threadIdx.x & 31;
  const int tile = blockIdx.x * WAVES_PER_BLOCK + wave;
  const int m = lane & 15, hi = lane >> 4, n = m;

  float*  in_tile = (float*) (smem + wave * WAVE_SMEM);
  half_t* h0      = (half_t*)(smem + wave * WAVE_SMEM + 6144);
  half_t* h1      = (half_t*)(smem + wave * WAVE_SMEM + 7168);
  half_t* merged  = (half_t*)(smem + wave * WAVE_SMEM + 8192);  // 16 x 64 f16

  // stage 0: stream input tile (16 rows x 96 f32) into per-wave LDS
  {
    const v4fa* g4 = (const v4fa*)(s_input + (size_t)tile * 16 * 96);
    v4fa* l4 = (v4fa*)in_tile;
    int row = lane >> 1, h2 = lane & 1;
#pragma unroll
    for (int i = 0; i < 12; ++i) {
      int c4 = h2 * 12 + i;
      l4[row * 24 + c4] = g4[row * 24 + c4];
    }
  }
  // zero merged K-pad columns 48..63
  {
    u32a* mz = (u32a*)merged;
#pragma unroll
    for (int i = 0; i < 4; ++i) {
      int idx = lane + 32 * i;  // 0..127
      mz[(idx >> 3) * 32 + 24 + (idx & 7)] = 0;
    }
  }
  comp_fence();

  // per-lane bias / LN params (column n of each layer)
  float b_en1a = en_b1[n], b_en1b = en_b1[n + 16], b_en2 = en_b2[n];
  float b_oa1a = oa_b1[n], b_oa1b = oa_b1[n + 16], b_oa2 = oa_b2[n];
  float b_g1a  = g_b1[n],  b_g1b  = g_b1[n + 16],  b_g2  = g_b2[n];
  float lng_o = oa_g[n], lnb_o = oa_bb[n], lng_g = gl_g[n], lnb_g = gl_bb[n];
  float b_a1a = a_b1[n], b_a1b = a_b1[n + 16];
  float b_a2a = a_b2[n], b_a2b = a_b2[n + 16];
  float b_a3 = (n < 2) ? a_b3[n] : 0.f;

  v8f z = {};
  v8f sc;  // self encoding, C-frag layout, live across both attention pools
  {
    v16h w1lo = load_bfrag(pk, 0, lane);
    v16h w1hi = load_bfrag(pk, 1, lane);
    v16h w2   = load_bfrag(pk, 2, lane);
    v4fa xs = *(const v4fa*)(in_tile + m * 96);
    v16h a1 = make_a1(xs[0], xs[1], xs[2], xs[3], hi);
    v8f c0 = wmma_f16(a1, w1lo, z);
    v8f c1 = wmma_f16(a1, w1hi, z);
    store_act(h0, 32, 0,  c0, b_en1a, false, lane);
    store_act(h0, 32, 16, c1, b_en1b, false, lane);
    v16h a2 = load_afrag(h0, 32, lane);
    sc = wmma_f16(a2, w2, z);
#pragma unroll
    for (int r = 0; r < 8; ++r) {
      float v = fmaxf(sc[r] + b_en2, 0.f);
      sc[r] = v;
      ((h16a*)merged)[(r + hi * 8) * 64 + n] = (half_t)v;  // cols 0..15 = self
    }
  }

  // other-agent pool -> merged cols 32..47 ; food pool -> merged cols 16..31
  attend_pool<15, true>(in_tile, h0, h1, merged, lane, pk, 3,
                        b_oa1a, b_oa1b, b_oa2, lng_o, lnb_o, 32, sc);
  attend_pool<16, false>(in_tile, h0, h1, merged, lane, pk, 6,
                         b_g1a, b_g1b, b_g2, lng_g, lnb_g, 16, sc);
  comp_fence();

  // action head: 48->32 (leaky) -> 32 (leaky) -> 2 (tanh)
  {
    v16h w1k0n0 = load_bfrag(pk, 9, lane);
    v16h w1k0n1 = load_bfrag(pk, 10, lane);
    v16h w1k1n0 = load_bfrag(pk, 11, lane);
    v16h w1k1n1 = load_bfrag(pk, 12, lane);
    v16h a1a = load_afrag(merged, 64, lane);       // K 0..31
    v16h a1b = load_afrag(merged + 32, 64, lane);  // K 32..63 (48..63 zero)
    v8f c0 = wmma_f16(a1a, w1k0n0, z);
    c0 = wmma_f16(a1b, w1k1n0, c0);
    v8f c1 = wmma_f16(a1a, w1k0n1, z);
    c1 = wmma_f16(a1b, w1k1n1, c1);
    store_act(h0, 32, 0,  c0, b_a1a, true, lane);
    store_act(h0, 32, 16, c1, b_a1b, true, lane);
    v16h a2 = load_afrag(h0, 32, lane);
    v16h w2n0 = load_bfrag(pk, 13, lane);
    v16h w2n1 = load_bfrag(pk, 14, lane);
    c0 = wmma_f16(a2, w2n0, z);
    c1 = wmma_f16(a2, w2n1, z);
    store_act(h1, 32, 0,  c0, b_a2a, true, lane);
    store_act(h1, 32, 16, c1, b_a2b, true, lane);
    v16h a3 = load_afrag(h1, 32, lane);
    v16h w3 = load_bfrag(pk, 15, lane);
    v8f cc = wmma_f16(a3, w3, z);
    if (n < 2) {  // lanes 0,1 (rows 0..7) and 16,17 (rows 8..15)
      float* op = out + (size_t)tile * 32;
#pragma unroll
      for (int r = 0; r < 8; ++r)
        op[(r + hi * 8) * 2 + n] = tanhf(cc[r] + b_a3);
    }
  }
}

extern "C" void kernel_launch(void* const* d_in, const int* in_sizes, int n_in,
                              void* d_out, int out_size, void* d_ws, size_t ws_size,
                              hipStream_t stream) {
  const float* s_input = (const float*)d_in[0];
  const float* en_W1 = (const float*)d_in[1];
  const float* en_b1 = (const float*)d_in[2];
  const float* en_W2 = (const float*)d_in[3];
  const float* en_b2 = (const float*)d_in[4];
  const float* oa_W1 = (const float*)d_in[5];
  const float* oa_b1 = (const float*)d_in[6];
  const float* oa_W2 = (const float*)d_in[7];
  const float* oa_b2 = (const float*)d_in[8];
  const float* g_W1  = (const float*)d_in[9];
  const float* g_b1  = (const float*)d_in[10];
  const float* g_W2  = (const float*)d_in[11];
  const float* g_b2  = (const float*)d_in[12];
  const float* oa_g  = (const float*)d_in[13];
  const float* oa_b  = (const float*)d_in[14];
  const float* gl_g  = (const float*)d_in[15];
  const float* gl_b  = (const float*)d_in[16];
  const float* a_W1  = (const float*)d_in[17];
  const float* a_b1  = (const float*)d_in[18];
  const float* a_W2  = (const float*)d_in[19];
  const float* a_b2  = (const float*)d_in[20];
  const float* a_W3  = (const float*)d_in[21];
  const float* a_b3  = (const float*)d_in[22];
  (void)n_in; (void)out_size; (void)ws_size;

  uint32_t* pk = (uint32_t*)d_ws;  // 16 frags * 256 dwords = 16 KB
  pack_weights<<<16, 32, 0, stream>>>(en_W1, en_W2, oa_W1, oa_W2, g_W1, g_W2,
                                      a_W1, a_W2, a_W3, pk);
  int nrows = in_sizes[0] / 96;  // B
  int blocks = nrows / (16 * WAVES_PER_BLOCK);
  actor_kernel<<<blocks, 128, 0, stream>>>(s_input, pk, en_b1, en_b2, oa_b1,
                                           oa_b2, g_b1, g_b2, oa_g, oa_b, gl_g,
                                           gl_b, a_b1, a_b2, a_b3,
                                           (float*)d_out);
}
